// EGNN_FSL_85298050498825
// MI455X (gfx1250) — matrix operand order repaired and checked by reading, not compiled, not run on hardware
//
#include <hip/hip_runtime.h>
#include <hip/hip_bf16.h>
#include <cmath>

// ---------------------------------------------------------------------------
// Types for CDNA5 WMMA (wave32): A/B = 16 halves per lane, C/D = 8 floats.
// ---------------------------------------------------------------------------
typedef _Float16 half16 __attribute__((ext_vector_type(16)));
typedef float   floatx8 __attribute__((ext_vector_type(8)));
typedef unsigned int uint32x4 __attribute__((ext_vector_type(4)));
typedef int     int32x4 __attribute__((ext_vector_type(4)));
typedef int     int32x8 __attribute__((ext_vector_type(8)));

union Frag {
  half16 h;
  uint4  u[2];
};
union H8 {
  uint4 u;
  _Float16 h[8];
};

static constexpr int Bsz       = 32;
static constexpr int Nn        = 100;
static constexpr int NF        = 128;
static constexpr int PAIRS     = Nn * Nn;        // 10000
static constexpr int PAIR_TILES = PAIRS / 16;    // 625 (exact)
static constexpr int EDGE_BLOCKS = (PAIR_TILES + 7) / 8;  // 79
static constexpr int NODE_ROWS = Bsz * Nn;       // 3200
static constexpr int NODE_TILES = NODE_ROWS / 16; // 200 (exact)
static constexpr float ALPHA = 0.3f;

__device__ __forceinline__ floatx8 wmma16(half16 a, half16 b, floatx8 c) {
  // D = A(16x32 f16) * B(32x16 f16) + C(16x16 f32)
  return __builtin_amdgcn_wmma_f32_16x16x32_f16(
      /*neg_a=*/false, a, /*neg_b=*/false, b,
      /*c_mod=*/(short)0, c, /*reuse_a=*/false, /*reuse_b=*/false);
}

__device__ __forceinline__ float leaky(float v) {
  return v > 0.0f ? v : ALPHA * v;
}

__device__ __forceinline__ float waveReduceSum(float v) {
  #pragma unroll
  for (int m = 16; m > 0; m >>= 1) v += __shfl_xor(v, m, 32);
  return v;
}

// ---------------------------------------------------------------------------
// Tensor Data Mover: 1-row tile DMA of `bytes` (multiple of 8) from global to
// LDS. D# per CDNA5 ISA ch.8: group0 {count=1, lds_addr, global_addr, type=2},
// group1 {data_size=8B, tensor_dim0=tile_dim0=bytes/8, tensor_dim1=1,
// tile_dim1=1, stride0=units}. Tracked by TENSORcnt.
// ---------------------------------------------------------------------------
__device__ __forceinline__ void tdmStage(void* ldsDst, const void* gsrc,
                                         unsigned bytes) {
  const unsigned units = bytes >> 3;  // 8-byte elements
  const unsigned ldsAddr = (unsigned)(uintptr_t)ldsDst;
  const unsigned long long ga = (unsigned long long)(uintptr_t)gsrc;

  uint32x4 g0 = {0u, 0u, 0u, 0u};
  g0[0] = 1u;                                   // count=1, user descriptor
  g0[1] = ldsAddr;                              // lds_addr [63:32]
  g0[2] = (unsigned)(ga & 0xffffffffu);         // global_addr [95:64]
  g0[3] = (unsigned)((ga >> 32) & 0x01ffffffu)  // global_addr [120:96]
          | (2u << 30);                         // type=2 ("image")

  int32x8 g1 = {0, 0, 0, 0, 0, 0, 0, 0};
  g1[0] = (int)(3u << 16);                      // workgroup_mask=0, data_size=8B
  g1[1] = (int)((units & 0xffffu) << 16);       // tensor_dim0[15:0]
  g1[2] = (int)((units >> 16) | (1u << 16));    // tensor_dim0[31:16], tensor_dim1=1
  g1[3] = (int)((units & 0xffffu) << 16);       // tile_dim0
  g1[4] = 1;                                    // tile_dim1=1 (single row)
  g1[5] = (int)units;                           // tensor_dim0_stride[31:0]
  // g1[6], g1[7] = 0

  int32x4 gz = {0, 0, 0, 0};
#if __clang_major__ >= 23
  int32x8 gz8 = {0, 0, 0, 0, 0, 0, 0, 0};
  __builtin_amdgcn_tensor_load_to_lds(g0, g1, gz, gz, gz8, 0);
#else
  __builtin_amdgcn_tensor_load_to_lds(g0, g1, gz, gz, 0);
#endif
}

// Build A-fragment (16-bit A 16x32 layout) from a row of f32:
//   h[t] = row[base0 + t] (t<8), row[base0+16+(t-8)] (t>=8),
// base0 = chunk*32 + hi*8.
__device__ __forceinline__ void buildRowFrag(Frag& f, const float* __restrict__ r,
                                             int base0) {
  #pragma unroll
  for (int g = 0; g < 2; ++g) {
    const int base = base0 + g * 16;
    float4 a0 = *(const float4*)(r + base);
    float4 a1 = *(const float4*)(r + base + 4);
    f.h[g * 8 + 0] = (_Float16)a0.x;  f.h[g * 8 + 1] = (_Float16)a0.y;
    f.h[g * 8 + 2] = (_Float16)a0.z;  f.h[g * 8 + 3] = (_Float16)a0.w;
    f.h[g * 8 + 4] = (_Float16)a1.x;  f.h[g * 8 + 5] = (_Float16)a1.y;
    f.h[g * 8 + 6] = (_Float16)a1.z;  f.h[g * 8 + 7] = (_Float16)a1.w;
  }
}

// Same but element-wise |ri - rj| (pairwise abs-diff features).
__device__ __forceinline__ void buildAbsDiffFrag(Frag& f,
                                                 const float* __restrict__ ri,
                                                 const float* __restrict__ rj,
                                                 int base0) {
  #pragma unroll
  for (int g = 0; g < 2; ++g) {
    const int base = base0 + g * 16;
    float4 a0 = *(const float4*)(ri + base);
    float4 a1 = *(const float4*)(ri + base + 4);
    float4 b0 = *(const float4*)(rj + base);
    float4 b1 = *(const float4*)(rj + base + 4);
    f.h[g * 8 + 0] = (_Float16)fabsf(a0.x - b0.x);
    f.h[g * 8 + 1] = (_Float16)fabsf(a0.y - b0.y);
    f.h[g * 8 + 2] = (_Float16)fabsf(a0.z - b0.z);
    f.h[g * 8 + 3] = (_Float16)fabsf(a0.w - b0.w);
    f.h[g * 8 + 4] = (_Float16)fabsf(a1.x - b1.x);
    f.h[g * 8 + 5] = (_Float16)fabsf(a1.y - b1.y);
    f.h[g * 8 + 6] = (_Float16)fabsf(a1.z - b1.z);
    f.h[g * 8 + 7] = (_Float16)fabsf(a1.w - b1.w);
  }
}

// Load an A-fragment from an f16 activation row held in LDS (pitch 256 halves).
__device__ __forceinline__ void loadActFrag(Frag& f,
                                            const _Float16* __restrict__ actRow,
                                            int base0) {
  f.u[0] = *(const uint4*)(actRow + base0);
  f.u[1] = *(const uint4*)(actRow + base0 + 16);
}

// Store one 16x16 f32 accumulator tile to the wave's LDS activation buffer
// with bias + LeakyReLU, converting to f16.
// D layout: VGPR r -> (M = r + 8*hi, N = nt*16 + (lane&15)).
__device__ __forceinline__ void storeActTile(_Float16* __restrict__ actW,
                                             floatx8 acc, int nt, int lane,
                                             const float* __restrict__ bias) {
  const int n  = nt * 16 + (lane & 15);
  const int hi = lane >> 4;
  const float bv = bias[n];
  #pragma unroll
  for (int r = 0; r < 8; ++r) {
    float v = leaky(acc[r] + bv);
    actW[(r + 8 * hi) * 256 + n] = (_Float16)v;
  }
}

// One WMMA MLP layer with B fragments resident in LDS (NK compile-time).
template <int NK>
__device__ __forceinline__ void mlpLayerLDS(const _Float16* __restrict__ wS,
                                            const Frag* __restrict__ a, int nT,
                                            _Float16* __restrict__ actW,
                                            int lane,
                                            const float* __restrict__ bias) {
  const half16* wf = (const half16*)wS;
  for (int nt = 0; nt < nT; ++nt) {
    floatx8 acc = {0, 0, 0, 0, 0, 0, 0, 0};
    #pragma unroll
    for (int c = 0; c < NK; ++c) {
      half16 bw = wf[(nt * NK + c) * 32 + lane];
      acc = wmma16(a[c].h, bw, acc);
    }
    storeActTile(actW, acc, nt, lane, bias);
  }
}

// ---------------------------------------------------------------------------
// Weight packing: fp32 [K,Nout] * scale[Nout] -> f16 in WMMA B-fragment order.
// B 32x16 16-bit layout: lanes 0-15 hold K=0..7,16..23 of the chunk, lanes
// 16-31 K=8..15,24..31. Fragment = 512 halves: ((nt*nK+kc)*32+lane)*16 + t.
// ---------------------------------------------------------------------------
__global__ void pack_weights_kernel(const float* __restrict__ W,
                                    const float* __restrict__ S,
                                    _Float16* __restrict__ dst, int K, int Nout) {
  const int flat = blockIdx.x * blockDim.x + threadIdx.x;
  if (flat >= K * Nout) return;
  const int t = flat & 15;
  const int lane = (flat >> 4) & 31;
  const int fragIdx = flat >> 9;
  const int nK = K >> 5;
  const int kc = fragIdx % nK;
  const int nt = fragIdx / nK;
  const int n = nt * 16 + (lane & 15);
  const int hi = lane >> 4;
  const int k = kc * 32 + hi * 8 + (t & 7) + ((t >> 3) << 4);
  dst[flat] = (_Float16)(W[(size_t)k * Nout + n] * S[n]);
}

// ---------------------------------------------------------------------------
// Node aggregation: L1-normalize masked edge row, aggregate neighbors, build
// x = [nf, aggr0, aggr1] in [B,N,384]. One wave per (b,e,i).
// ---------------------------------------------------------------------------
__global__ void node_aggr_kernel(const float* __restrict__ nf,
                                 const float* __restrict__ ef,
                                 float* __restrict__ x) {
  const int i = blockIdx.x, e = blockIdx.y, b = blockIdx.z;
  const int lane = threadIdx.x;
  __shared__ float efn[128];

  const float* erow = ef + (((size_t)b * 2 + e) * Nn + i) * Nn;
  float vals[4];
  float s = 0.0f;
  #pragma unroll
  for (int t = 0; t < 4; ++t) {
    const int j = lane + 32 * t;
    float v = 0.0f;
    if (j < Nn && j != i) v = erow[j];
    vals[t] = v;
    s += fabsf(v);
  }
  s = waveReduceSum(s);
  const float inv = (s == 0.0f) ? 0.0f : 1.0f / s;
  #pragma unroll
  for (int t = 0; t < 4; ++t) efn[lane + 32 * t] = vals[t] * inv;
  __syncthreads();

  const float* nfb = nf + (size_t)b * Nn * NF;
  float* xr = x + ((size_t)b * Nn + i) * 384;
  #pragma unroll
  for (int t = 0; t < 4; ++t) {
    const int d = lane + 32 * t;
    float acc = 0.0f;
    for (int j = 0; j < Nn; ++j) acc += efn[j] * nfb[j * NF + d];
    xr[NF + e * NF + d] = acc;
    if (e == 0) xr[d] = nfb[i * NF + d];
  }
}

// ---------------------------------------------------------------------------
// Node MLP: [3200,384] -> 256 -> 128 via WMMA. 8 waves/block, 1 M-tile/wave.
// B fragments streamed from L2 (weights tiny); D->A shuffle via per-wave LDS.
// ---------------------------------------------------------------------------
__global__ void node_mlp_kernel(const float* __restrict__ x,
                                const _Float16* __restrict__ W0p,
                                const float* __restrict__ B0,
                                const _Float16* __restrict__ W1p,
                                const float* __restrict__ B1,
                                float* __restrict__ nfOut) {
  extern __shared__ char smem[];
  _Float16* actS = (_Float16*)smem;  // 8 waves * 16 rows * 256 halves = 64KB

  const int tid = threadIdx.x;
  const int wave = tid >> 5, lane = tid & 31;
  const int tile = blockIdx.x * 8 + wave;  // 200 tiles, all valid
  const int m = lane & 15, hi = lane >> 4;
  const int row = tile * 16 + m;
  const float* xr = x + (size_t)row * 384;
  _Float16* actW = actS + wave * 16 * 256;

  __builtin_prefetch(W0p + (size_t)tid * 768, 0, 1);

  Frag a[12];
  #pragma unroll
  for (int c = 0; c < 12; ++c) buildRowFrag(a[c], xr, c * 32 + hi * 8);

  // layer 1: 384 -> 256
  const half16* w0 = (const half16*)W0p;
  for (int nt = 0; nt < 16; ++nt) {
    floatx8 acc = {0, 0, 0, 0, 0, 0, 0, 0};
    #pragma unroll
    for (int c = 0; c < 12; ++c) {
      half16 bw = w0[(nt * 12 + c) * 32 + lane];
      acc = wmma16(a[c].h, bw, acc);
    }
    storeActTile(actW, acc, nt, lane, B0);
  }

  Frag a1[8];
  #pragma unroll
  for (int c = 0; c < 8; ++c) loadActFrag(a1[c], actW + m * 256, c * 32 + hi * 8);

  // layer 2: 256 -> 128, write straight to global nf.
  const half16* w1 = (const half16*)W1p;
  for (int nt = 0; nt < 8; ++nt) {
    floatx8 acc = {0, 0, 0, 0, 0, 0, 0, 0};
    #pragma unroll
    for (int c = 0; c < 8; ++c) {
      half16 bw = w1[(nt * 8 + c) * 32 + lane];
      acc = wmma16(a1[c].h, bw, acc);
    }
    const int n = nt * 16 + (lane & 15);
    const float bv = B1[n];
    #pragma unroll
    for (int r = 0; r < 8; ++r) {
      const int gr = tile * 16 + r + 8 * hi;
      nfOut[(size_t)gr * NF + n] = leaky(acc[r] + bv);
    }
  }
}

// ---------------------------------------------------------------------------
// Edge similarity MLP (hot kernel, ~95 GFLOP/layer):
// per 16-pair tile: h=|nf_i-nf_j| -> 256 -> 256 -> 128 -> 128 -> sigmoid(dot).
// 8 waves/block; weights DMA'd layer-by-layer into double-buffered LDS by the
// Tensor Data Mover (wave 0, TENSORcnt), overlapping the next layer's weight
// transfer with the current layer's WMMAs. Per-wave 16x256 f16 LDS activation
// buffer performs the D-layout -> A-layout cross-lane shuffle.
// LDS: buf0 64KB @0 | buf1 128KB @64KB | act 64KB @192KB  (256KB total).
// ---------------------------------------------------------------------------
__global__ void edge_mlp_kernel(const float* __restrict__ nf,
                                const _Float16* __restrict__ W0p,
                                const _Float16* __restrict__ W1p,
                                const _Float16* __restrict__ W2p,
                                const _Float16* __restrict__ W3p,
                                const float* __restrict__ B0,
                                const float* __restrict__ B1,
                                const float* __restrict__ B2,
                                const float* __restrict__ B3,
                                const float* __restrict__ Wout,
                                const float* __restrict__ boutp,
                                float* __restrict__ sim) {
  extern __shared__ char smem[];
  _Float16* wS0  = (_Float16*)smem;             // 64KB  (W0, then W2)
  _Float16* wS1  = (_Float16*)(smem + 65536);   // 128KB (W1, then W3)
  _Float16* actS = (_Float16*)(smem + 196608);  // 64KB activations

  const int tid = threadIdx.x;
  const int wave = tid >> 5, lane = tid & 31;
  const int b = blockIdx.y;
  int tile = blockIdx.x * 8 + wave;
  const bool valid = tile < PAIR_TILES;
  if (!valid) tile = PAIR_TILES - 1;  // clamp: uniform barriers, skip store

  const int m = lane & 15, hi = lane >> 4;
  const int p = tile * 16 + m;
  const int pi = p / Nn, pj = p % Nn;
  const float* ri = nf + ((size_t)b * Nn + pi) * NF;
  const float* rj = nf + ((size_t)b * Nn + pj) * NF;
  _Float16* actW = actS + wave * 16 * 256;

  // DMA layer-1 weights while every wave builds its A0 fragments.
  if (wave == 0) tdmStage(wS0, W0p, 4 * 16 * 512 * 2);

  Frag a[8];
  #pragma unroll
  for (int c = 0; c < 4; ++c) buildAbsDiffFrag(a[c], ri, rj, c * 32 + hi * 8);

  if (wave == 0) __builtin_amdgcn_s_wait_tensorcnt(0);
  __syncthreads();                                 // W0 visible in wS0
  if (wave == 0) tdmStage(wS1, W1p, 8 * 16 * 512 * 2);  // overlap with L1

  // layer 1: 128 -> 256 (from wS0)
  mlpLayerLDS<4>(wS0, a, 16, actW, lane, B0);
  __syncthreads();                                 // all done reading wS0
  #pragma unroll
  for (int c = 0; c < 8; ++c) loadActFrag(a[c], actW + m * 256, c * 32 + hi * 8);
  if (wave == 0) __builtin_amdgcn_s_wait_tensorcnt(0);
  __syncthreads();                                 // W1 visible in wS1
  if (wave == 0) tdmStage(wS0, W2p, 8 * 8 * 512 * 2);   // overlap with L2

  // layer 2: 256 -> 256 (from wS1)
  mlpLayerLDS<8>(wS1, a, 16, actW, lane, B1);
  __syncthreads();                                 // all done reading wS1
  #pragma unroll
  for (int c = 0; c < 8; ++c) loadActFrag(a[c], actW + m * 256, c * 32 + hi * 8);
  if (wave == 0) __builtin_amdgcn_s_wait_tensorcnt(0);
  __syncthreads();                                 // W2 visible in wS0
  if (wave == 0) tdmStage(wS1, W3p, 4 * 8 * 512 * 2);   // overlap with L3

  // layer 3: 256 -> 128 (from wS0)
  mlpLayerLDS<8>(wS0, a, 8, actW, lane, B2);
  __syncthreads();
  #pragma unroll
  for (int c = 0; c < 4; ++c) loadActFrag(a[c], actW + m * 256, c * 32 + hi * 8);
  if (wave == 0) __builtin_amdgcn_s_wait_tensorcnt(0);
  __syncthreads();                                 // W3 visible in wS1

  // layer 4: 128 -> 128 (from wS1)
  mlpLayerLDS<4>(wS1, a, 8, actW, lane, B3);

  // final dot + sigmoid (lanes 0..15 own one row each; hi half duplicates)
  const float bout = boutp[0];
  const _Float16* hrow = actW + (lane & 15) * 256;
  float dot = 0.0f;
  #pragma unroll
  for (int k = 0; k < NF; k += 8) {
    H8 xv;
    xv.u = *(const uint4*)(hrow + k);
    #pragma unroll
    for (int q = 0; q < 8; ++q) dot += (float)xv.h[q] * Wout[k + q];
  }
  const float s = 1.0f / (1.0f + __expf(-(dot + bout)));
  if (valid && lane < 16) {
    sim[(size_t)b * PAIRS + tile * 16 + lane] = s;
  }
}

// ---------------------------------------------------------------------------
// Edge finalize: masked ef, merge_sum, sim-weighted L1 renorm, force diag,
// normalize across the 2 edge channels. One wave per (b,i) row.
// ---------------------------------------------------------------------------
__global__ void edge_finalize_kernel(const float* __restrict__ ef,
                                     const float* __restrict__ sim,
                                     float* __restrict__ out) {
  const int i = blockIdx.x, b = blockIdx.y, lane = threadIdx.x;
  const float* e0r = ef + (((size_t)b * 2 + 0) * Nn + i) * Nn;
  const float* e1r = ef + (((size_t)b * 2 + 1) * Nn + i) * Nn;
  const float* sr  = sim + ((size_t)b * Nn + i) * Nn;

  float v0[4], v1[4];
  float ms0 = 0, ms1 = 0, n0 = 0, n1 = 0;
  #pragma unroll
  for (int t = 0; t < 4; ++t) {
    const int j = lane + 32 * t;
    const bool act = (j < Nn);
    const float a0 = (act && j != i) ? e0r[j] : 0.0f;
    const float a1 = (act && j != i) ? e1r[j] : 0.0f;
    const float s  = act ? sr[j] : 0.0f;
    ms0 += a0;
    ms1 += a1;
    v0[t] = a0 * s;
    v1[t] = a1 * (1.0f - s);
    n0 += fabsf(v0[t]);
    n1 += fabsf(v1[t]);
  }
  ms0 = waveReduceSum(ms0);
  ms1 = waveReduceSum(ms1);
  n0 = waveReduceSum(n0);
  n1 = waveReduceSum(n1);
  const float i0 = (n0 == 0.0f) ? 0.0f : 1.0f / n0;
  const float i1 = (n1 == 0.0f) ? 0.0f : 1.0f / n1;

  #pragma unroll
  for (int t = 0; t < 4; ++t) {
    const int j = lane + 32 * t;
    if (j < Nn) {
      float r0 = v0[t] * i0 * ms0 + ((j == i) ? 1.0f : 0.0f) + 1e-6f;
      float r1 = v1[t] * i1 * ms1 + 1e-6f;
      const float d = r0 + r1;
      out[(((size_t)b * 2 + 0) * Nn + i) * Nn + j] = r0 / d;
      out[(((size_t)b * 2 + 1) * Nn + i) * Nn + j] = r1 / d;
    }
  }
}

// ---------------------------------------------------------------------------
// Host: workspace layout, weight packing, 3-layer pipeline.
// ---------------------------------------------------------------------------
extern "C" void kernel_launch(void* const* d_in, const int* in_sizes, int n_in,
                              void* d_out, int out_size, void* d_ws,
                              size_t ws_size, hipStream_t stream) {
  (void)in_sizes; (void)n_in; (void)out_size; (void)ws_size;

  const float* nf_in = (const float*)d_in[0];
  const float* ef_in = (const float*)d_in[1];
  const float* nW0 = (const float*)d_in[2];
  const float* nS0 = (const float*)d_in[3];
  const float* nB0 = (const float*)d_in[4];
  const float* nW1 = (const float*)d_in[5];
  const float* nS1 = (const float*)d_in[6];
  const float* nB1 = (const float*)d_in[7];
  const float* eW0 = (const float*)d_in[8];
  const float* eS0 = (const float*)d_in[9];
  const float* eB0 = (const float*)d_in[10];
  const float* eW1 = (const float*)d_in[11];
  const float* eS1 = (const float*)d_in[12];
  const float* eB1 = (const float*)d_in[13];
  const float* eW2 = (const float*)d_in[14];
  const float* eS2 = (const float*)d_in[15];
  const float* eB2 = (const float*)d_in[16];
  const float* eW3 = (const float*)d_in[17];
  const float* eS3 = (const float*)d_in[18];
  const float* eB3 = (const float*)d_in[19];
  const float* eWout = (const float*)d_in[20];
  const float* eBout = (const float*)d_in[21];

  char* wptr = (char*)d_ws;
  auto alloc = [&](size_t bytes) -> char* {
    char* pRet = wptr;
    wptr += (bytes + 255) & ~(size_t)255;
    return pRet;
  };
  float* nfA = (float*)alloc((size_t)Bsz * Nn * NF * 4);
  float* nfB = (float*)alloc((size_t)Bsz * Nn * NF * 4);
  float* efA = (float*)alloc((size_t)Bsz * 2 * Nn * Nn * 4);
  float* efB = (float*)alloc((size_t)Bsz * 2 * Nn * Nn * 4);
  float* xbuf = (float*)alloc((size_t)NODE_ROWS * 384 * 4);
  float* simbuf = (float*)alloc((size_t)Bsz * PAIRS * 4);
  // Packed f16 weights: per layer 278528 halves.
  static constexpr size_t PK_NW0 = 0;           // 384*256 = 98304
  static constexpr size_t PK_NW1 = 98304;       // 256*128 = 32768
  static constexpr size_t PK_EW0 = 131072;      // 128*256 = 32768
  static constexpr size_t PK_EW1 = 163840;      // 256*256 = 65536
  static constexpr size_t PK_EW2 = 229376;      // 256*128 = 32768
  static constexpr size_t PK_EW3 = 262144;      // 128*128 = 16384
  static constexpr size_t PK_PER_LAYER = 278528;
  _Float16* pack = (_Float16*)alloc(3 * PK_PER_LAYER * sizeof(_Float16));

  // Copy initial node/edge features into ping-pong buffers.
  hipMemcpyAsync(nfA, nf_in, (size_t)Bsz * Nn * NF * 4,
                 hipMemcpyDeviceToDevice, stream);
  hipMemcpyAsync(efA, ef_in, (size_t)Bsz * 2 * Nn * Nn * 4,
                 hipMemcpyDeviceToDevice, stream);

  // Opt-in to large dynamic LDS (320KB/WGP on gfx1250).
  (void)hipFuncSetAttribute(reinterpret_cast<const void*>(edge_mlp_kernel),
                            hipFuncAttributeMaxDynamicSharedMemorySize, 262144);
  (void)hipFuncSetAttribute(reinterpret_cast<const void*>(node_mlp_kernel),
                            hipFuncAttributeMaxDynamicSharedMemorySize, 65536);

  auto packW = [&](const float* Wsrc, const float* Ssrc, _Float16* dst, int K,
                   int Nout) {
    const int total = K * Nout;  // always multiple of 256 here
    pack_weights_kernel<<<dim3(total / 256), dim3(256), 0, stream>>>(
        Wsrc, Ssrc, dst, K, Nout);
  };
  for (int l = 0; l < 3; ++l) {
    _Float16* base = pack + (size_t)l * PK_PER_LAYER;
    packW(nW0 + (size_t)l * 384 * 256, nS0 + l * 256, base + PK_NW0, 384, 256);
    packW(nW1 + (size_t)l * 256 * 128, nS1 + l * 128, base + PK_NW1, 256, 128);
    packW(eW0 + (size_t)l * 128 * 256, eS0 + l * 256, base + PK_EW0, 128, 256);
    packW(eW1 + (size_t)l * 256 * 256, eS1 + l * 256, base + PK_EW1, 256, 256);
    packW(eW2 + (size_t)l * 256 * 128, eS2 + l * 128, base + PK_EW2, 256, 128);
    packW(eW3 + (size_t)l * 128 * 128, eS3 + l * 128, base + PK_EW3, 128, 128);
  }

  float* nf_cur = nfA;
  float* nf_nxt = nfB;
  float* ef_cur = efA;
  float* ef_nxt = efB;
  for (int l = 0; l < 3; ++l) {
    const _Float16* base = pack + (size_t)l * PK_PER_LAYER;

    node_aggr_kernel<<<dim3(Nn, 2, Bsz), dim3(32), 0, stream>>>(
        nf_cur, ef_cur, xbuf);

    node_mlp_kernel<<<dim3(NODE_TILES / 8), dim3(256), 65536, stream>>>(
        xbuf, base + PK_NW0, nB0 + l * 256, base + PK_NW1, nB1 + l * 128,
        nf_nxt);

    edge_mlp_kernel<<<dim3(EDGE_BLOCKS, Bsz), dim3(256), 262144, stream>>>(
        nf_nxt, base + PK_EW0, base + PK_EW1, base + PK_EW2, base + PK_EW3,
        eB0 + l * 256, eB1 + l * 256, eB2 + l * 128, eB3 + l * 128,
        eWout + l * 128, eBout + l, simbuf);

    float* target = (l == 2) ? (float*)d_out : ef_nxt;
    edge_finalize_kernel<<<dim3(Nn, Bsz), dim3(32), 0, stream>>>(
        ef_cur, simbuf, target);

    // swap ping-pong
    float* tn = nf_cur; nf_cur = nf_nxt; nf_nxt = tn;
    if (l < 2) { float* te = ef_cur; ef_cur = ef_nxt; ef_nxt = te; }
  }
}
